// XTransEncoderAct_79963701117003
// MI455X (gfx1250) — matrix operand order repaired and checked by reading, not compiled, not run on hardware
//
#include <hip/hip_runtime.h>
#include <hip/hip_bf16.h>
#include <math.h>
#include <stdint.h>

typedef _Float16 half_t;
typedef __attribute__((ext_vector_type(16))) _Float16 v16h;
typedef __attribute__((ext_vector_type(8)))  _Float16 v8h;
typedef __attribute__((ext_vector_type(8)))  float    v8f;

#define TOK    4096        // B*N tokens
#define TOK3   12288       // B*N*3 vector rows
#define NSEQ   1024
#define NHEAD  16
#define DHID   512
#define FACTOR_F 0.08838834764831845f   // 0.5/sqrt(32)

// ---------------------------------------------------------------------------
// WMMA helpers (CDNA5 gfx1250, wave32)
// ---------------------------------------------------------------------------
__device__ __forceinline__ v8f wmma16(v16h a, v16h b, v8f c) {
  // D = A(16x32 f16) * B(32x16 f16) + C(16x16 f32)
  return __builtin_amdgcn_wmma_f32_16x16x32_f16(false, a, false, b, (short)0, c,
                                                false, false);
}

// A fragment: per-lane row = row0 + (lane&15); halves cover K = koff..koff+7 and
// koff+16..koff+23 with koff = (lane<16 ? 0 : 8).   (ISA 7.12.2, 16-bit A 16x32)
__device__ __forceinline__ v16h load_frag_a(const half_t* base, int row0, int ld, int k0) {
  const int lane = threadIdx.x & 31;
  const int row  = row0 + (lane & 15);
  const int koff = (lane < 16) ? 0 : 8;
  const half_t* p = base + (size_t)row * ld + k0 + koff;
  v8h lo = *reinterpret_cast<const v8h*>(p);
  v8h hi = *reinterpret_cast<const v8h*>(p + 16);
  v16h r;
#pragma unroll
  for (int i = 0; i < 8; ++i) { r[i] = lo[i]; r[8 + i] = hi[i]; }
  return r;
}

// B fragment from Bt stored N x K row-major (row = column of logical B):
// per-lane column = col0 + (lane&15); halves cover K = koff..koff+7 and
// koff+16..koff+23 with koff = (lane<16 ? 0 : 8).   Two 16B loads (works for global and LDS bases).
__device__ __forceinline__ v16h load_frag_b(const half_t* base, int col0, int ld, int k0) {
  const int lane = threadIdx.x & 31;
  const int col  = col0 + (lane & 15);
  const int koff = (lane < 16) ? 0 : 8;
  const half_t* p = base + (size_t)col * ld + k0 + koff;
  v8h lo = *reinterpret_cast<const v8h*>(p);
  v8h hi = *reinterpret_cast<const v8h*>(p + 16);
  v16h r;
#pragma unroll
  for (int i = 0; i < 8; ++i) { r[i] = lo[i]; r[8 + i] = hi[i]; }
  return r;
}

// ---------------------------------------------------------------------------
// Generic f16 WMMA GEMM: C(MxN,f32) = A(MxK,f16) @ Bt(NxK,f16)^T [+bias][+resid][silu]
// Block: 256 thr = 8 waves; wave tile 32x64; block tile 128(M) x 128(N).
// B tile (128 cols x 32 k, 8 KiB) is staged to LDS with CDNA5 async
// global->LDS copies (ASYNCcnt), double buffered to overlap the next tile's
// fetch with the current tile's WMMAs.
// Requires M%128==0, N%128==0, K%32==0 (true for every call here).
// ---------------------------------------------------------------------------
#define BS_LDW 40   // padded LDS row stride in halves (16B aligned, bank-spread)

__global__ void __launch_bounds__(256)
gemm_wmma_kernel(const half_t* __restrict__ A, int lda,
                 const half_t* __restrict__ Bt, int ldb,
                 const float* __restrict__ bias,
                 const float* __restrict__ resid, int ldr,
                 float* __restrict__ Cf, int ldcf,
                 half_t* __restrict__ Ch, int ldch,
                 int M, int N, int K, int act) {
  const int tid  = threadIdx.x;
  const int w    = tid >> 5;
  const int lane = tid & 31;
  const int m0  = blockIdx.y * 128 + (w >> 1) * 32;
  const int n0b = blockIdx.x * 128;              // block's N origin
  const int n0  = n0b + (w & 1) * 64;            // wave's N origin

  __shared__ __align__(16) half_t Bs[2][128][BS_LDW];

  // Async-stage one 128x32 B tile (as 512 16-byte chunks, 2 per thread).
  auto stage = [&](int buf, int k) {
#pragma unroll
    for (int c = tid; c < 512; c += 256) {
      const int row = c >> 2;
      const int seg = c & 3;
      const half_t* gp = Bt + (size_t)(n0b + row) * ldb + k + seg * 8;
      const uint32_t lo = (uint32_t)(uintptr_t)(&Bs[buf][row][seg * 8]);
      asm volatile("global_load_async_to_lds_b128 %0, %1, off"
                   :: "v"(lo), "v"((uint64_t)(uintptr_t)gp)
                   : "memory");
    }
  };

  v8f acc[2][4];
#pragma unroll
  for (int i = 0; i < 2; ++i)
#pragma unroll
    for (int f = 0; f < 4; ++f) acc[i][f] = {};

  stage(0, 0);
  for (int k = 0; k < K; k += 32) {
    const int  buf      = (k >> 5) & 1;
    const bool has_next = (k + 32 < K);
    if (has_next) {
      stage(buf ^ 1, k + 32);
      asm volatile("s_wait_asynccnt 2" ::: "memory");   // current tile done
    } else {
      asm volatile("s_wait_asynccnt 0" ::: "memory");
    }
    __syncthreads();   // all waves' chunks of this tile visible

    v16h a0 = load_frag_a(A, m0,      lda, k);
    v16h a1 = load_frag_a(A, m0 + 16, lda, k);
    const half_t* Bbase = &Bs[buf][0][0];
#pragma unroll
    for (int f = 0; f < 4; ++f) {
      v16h b = load_frag_b(Bbase, (w & 1) * 64 + f * 16, BS_LDW, 0);
      acc[0][f] = wmma16(a0, b, acc[0][f]);
      acc[1][f] = wmma16(a1, b, acc[1][f]);
    }
    __syncthreads();   // reads of buf done before it is restaged
  }

#pragma unroll
  for (int i = 0; i < 2; ++i) {
#pragma unroll
    for (int f = 0; f < 4; ++f) {
      const int n  = n0 + f * 16 + (lane & 15);
      const float bv = bias ? bias[n] : 0.0f;
#pragma unroll
      for (int r = 0; r < 8; ++r) {
        const int m = m0 + i * 16 + r + ((lane >= 16) ? 8 : 0);
        float v = acc[i][f][r] + bv;
        if (resid) v += resid[(size_t)m * ldr + n];
        if (act == 1) v = v / (1.0f + __expf(-v));   // silu
        if (Cf) Cf[(size_t)m * ldcf + n] = v;
        if (Ch) Ch[(size_t)m * ldch + n] = (half_t)v;
      }
    }
  }
}

// ---------------------------------------------------------------------------
// Flash-style masked-bias attention.
// grid.x = B * NHEAD * (NSEQ/128); block = 256 thr = 8 waves; 16 n-rows / wave.
// Hq/Hk: (B,N,16,128) f16 row-major.  Vt: (B,16,128,N) f16 (V_attn transposed).
// Writes resH (TOK x 512) f16 and resV (TOK3 x 512) f16 for the Wo/Wvo GEMMs.
// ---------------------------------------------------------------------------
__global__ void __launch_bounds__(256)
attn_kernel(const half_t* __restrict__ Hq, const half_t* __restrict__ Hk,
            const half_t* __restrict__ Vt,
            const float* __restrict__ rbf, const float* __restrict__ Db,
            const int* __restrict__ mask,
            half_t* __restrict__ resH, half_t* __restrict__ resV) {
  const int w    = threadIdx.x >> 5;
  const int lane = threadIdx.x & 31;
  int t = blockIdx.x;
  const int ntile = t & 7;   t >>= 3;
  const int h     = t & 15;  t >>= 4;
  const int b     = t;
  const int n0 = ntile * 128 + w * 16;

  const half_t* HqB = Hq + (size_t)b * NSEQ * 2048 + h * 128;
  const half_t* HkB = Hk + (size_t)b * NSEQ * 2048 + h * 128;
  const half_t* VtB = Vt + ((size_t)(b * NHEAD + h) * 128) * NSEQ;
  const float*  rbfB = rbf + (size_t)b * NSEQ * NSEQ;
  const float*  DbB  = Db  + (size_t)b * NSEQ * NSEQ;
  const int*    maskB = mask + b * NSEQ;

  const float NEG_INF = -__builtin_inff();

  v16h q[4];
#pragma unroll
  for (int kd = 0; kd < 4; ++kd) q[kd] = load_frag_a(HqB, n0, 2048, kd * 32);

  v8f acc[8];
#pragma unroll
  for (int f = 0; f < 8; ++f) acc[f] = {};
  float run_max[8], run_sum[8];
#pragma unroll
  for (int r = 0; r < 8; ++r) { run_max[r] = NEG_INF; run_sum[r] = 0.0f; }

  __shared__ __align__(16) half_t Plds[8][16][32];

  for (int m0 = 0; m0 < NSEQ; m0 += 32) {
    // ---- S = Q @ K^T for 16 rows x 32 keys ----
    v8f s[2]; s[0] = {}; s[1] = {};
#pragma unroll
    for (int j = 0; j < 2; ++j)
#pragma unroll
      for (int kd = 0; kd < 4; ++kd) {
        v16h kf = load_frag_b(HkB, m0 + j * 16, 2048, kd * 32);
        s[j] = wmma16(q[kd], kf, s[j]);
      }

    // ---- bias + mask in C-fragment layout ----
    const int mcol  = m0 + (lane & 15);
    const int rbase = n0 + ((lane >= 16) ? 8 : 0);
#pragma unroll
    for (int j = 0; j < 2; ++j) {
      const int m = mcol + j * 16;
      const bool keep = (maskB[m] != 0);
#pragma unroll
      for (int r = 0; r < 8; ++r) {
        const int n = rbase + r;
        float v = s[j][r] * FACTOR_F + rbfB[(size_t)n * NSEQ + m] +
                  DbB[(size_t)n * NSEQ + m];
        s[j][r] = keep ? v : NEG_INF;
      }
    }

    // ---- online softmax per row (rows live in 16-lane groups) ----
#pragma unroll
    for (int r = 0; r < 8; ++r) {
      float mx = fmaxf(s[0][r], s[1][r]);
#pragma unroll
      for (int off = 1; off <= 8; off <<= 1) mx = fmaxf(mx, __shfl_xor(mx, off, 32));
      const float nm = fmaxf(run_max[r], mx);
      const float corr = (run_max[r] == NEG_INF) ? 0.0f : __expf(run_max[r] - nm);
      float p0 = (nm == NEG_INF) ? 0.0f : __expf(s[0][r] - nm);
      float p1 = (nm == NEG_INF) ? 0.0f : __expf(s[1][r] - nm);
      s[0][r] = p0; s[1][r] = p1;
      float ts = p0 + p1;
#pragma unroll
      for (int off = 1; off <= 8; off <<= 1) ts += __shfl_xor(ts, off, 32);
      run_sum[r] = run_sum[r] * corr + ts;
      run_max[r] = nm;
#pragma unroll
      for (int f = 0; f < 8; ++f) acc[f][r] *= corr;
    }

    // ---- transpose P (C-layout) -> A-layout via per-wave LDS tile ----
#pragma unroll
    for (int r = 0; r < 8; ++r) {
      const int row = r + ((lane >= 16) ? 8 : 0);
      Plds[w][row][(lane & 15)]      = (half_t)s[0][r];
      Plds[w][row][16 + (lane & 15)] = (half_t)s[1][r];
    }
    __syncthreads();   // uniform trip count across all 8 waves

    v16h pa = load_frag_a(&Plds[w][0][0], 0, 32, 0);
#pragma unroll
    for (int f = 0; f < 8; ++f) {
      v16h vb = load_frag_b(VtB, f * 16, NSEQ, m0);
      acc[f] = wmma16(pa, vb, acc[f]);
    }
    __syncthreads();
  }

  // ---- normalize and scatter into resH / resV (GEMM-ready f16 layouts) ----
#pragma unroll
  for (int r = 0; r < 8; ++r) {
    const int n = n0 + r + ((lane >= 16) ? 8 : 0);
    const float inv = 1.0f / run_sum[r];
    const size_t tok = (size_t)b * NSEQ + n;
#pragma unroll
    for (int f = 0; f < 8; ++f) {
      const int e = f * 16 + (lane & 15);
      const float val = acc[f][r] * inv;
      if (e < 32) {
        resH[tok * 512 + h * 32 + e] = (half_t)val;
      } else {
        const int c = (e - 32) >> 5, e2 = (e - 32) & 31;
        resV[(tok * 3 + c) * 512 + h * 32 + e2] = (half_t)val;
      }
    }
  }
}

// ---------------------------------------------------------------------------
// Elementwise / staging kernels
// ---------------------------------------------------------------------------
__global__ void __launch_bounds__(256)
layernorm_kernel(const float* __restrict__ X, const float* __restrict__ g,
                 const float* __restrict__ be, half_t* __restrict__ Y, int D) {
  const int w = threadIdx.x >> 5, lane = threadIdx.x & 31;
  const size_t row = (size_t)blockIdx.x * 8 + w;
  const float* x = X + row * D;
  float s = 0.0f, s2 = 0.0f;
  for (int i = lane; i < D; i += 32) { float v = x[i]; s += v; s2 += v * v; }
#pragma unroll
  for (int off = 1; off < 32; off <<= 1) {
    s  += __shfl_xor(s, off, 32);
    s2 += __shfl_xor(s2, off, 32);
  }
  const float mean = s / D;
  const float var  = s2 / D - mean * mean;
  const float rstd = rsqrtf(var + 1e-5f);
  half_t* y = Y + row * D;
  for (int i = lane; i < D; i += 32)
    y[i] = (half_t)((x[i] - mean) * rstd * g[i] + be[i]);
}

__global__ void __launch_bounds__(256)
cast_kernel(const float* __restrict__ in, half_t* __restrict__ out, long n) {
  long i = (long)blockIdx.x * blockDim.x + threadIdx.x;
  const long st = (long)gridDim.x * blockDim.x;
  for (; i < n; i += st) out[i] = (half_t)in[i];
}

// W (K x N, f32 row-major) -> Wt (N x K, f16 row-major)
__global__ void __launch_bounds__(256)
transpose_cast_kernel(const float* __restrict__ W, half_t* __restrict__ Wt,
                      int K, int N) {
  long i = (long)blockIdx.x * blockDim.x + threadIdx.x;
  const long n = (long)K * N;
  const long st = (long)gridDim.x * blockDim.x;
  for (; i < n; i += st) {
    const int k = (int)(i / N), c = (int)(i % N);
    Wt[(size_t)c * K + k] = (half_t)W[i];
  }
}

// Build V_attn^T : (B,16,128,N) f16 from Hv (TOKx512) and Vv (TOK3x512)
__global__ void __launch_bounds__(256)
pack_vt_kernel(const half_t* __restrict__ Hv, const half_t* __restrict__ Vv,
               half_t* __restrict__ Vt) {
  long i = (long)blockIdx.x * blockDim.x + threadIdx.x;
  const long total = (long)4 * NHEAD * 128 * NSEQ;   // 8388608
  const long st = (long)gridDim.x * blockDim.x;
  for (; i < total; i += st) {
    long x = i;
    const int n = (int)(x & (NSEQ - 1)); x >>= 10;
    const int e = (int)(x & 127);        x >>= 7;
    const int h = (int)(x & 15);         x >>= 4;
    const int b = (int)x;
    const size_t tok = (size_t)b * NSEQ + n;
    half_t v;
    if (e < 32) {
      v = Hv[tok * 512 + h * 32 + e];
    } else {
      const int c = (e - 32) >> 5, e2 = (e - 32) & 31;
      v = Vv[(tok * 3 + c) * 512 + h * 32 + e2];
    }
    Vt[i] = v;   // i == ((b*16+h)*128+e)*NSEQ + n
  }
}

// scaler (TOK x 1024) f16 = concat( Hn2 , sqrt(sum_c V1^2) )
__global__ void __launch_bounds__(256)
pack_scaler_kernel(const half_t* __restrict__ Hn2, const float* __restrict__ Vp,
                   half_t* __restrict__ scaler) {
  long i = (long)blockIdx.x * blockDim.x + threadIdx.x;
  const long total = (long)TOK * 1024;
  const long st = (long)gridDim.x * blockDim.x;
  for (; i < total; i += st) {
    const int  j = (int)(i & 1023);
    const size_t t = (size_t)(i >> 10);
    if (j < 512) {
      scaler[i] = Hn2[t * 512 + j];
    } else {
      const int j2 = j - 512;
      float ss = 0.0f;
#pragma unroll
      for (int c = 0; c < 3; ++c) {
        const float v = Vp[(t * 3 + c) * 1024 + j2];
        ss += v * v;
      }
      scaler[i] = (half_t)sqrtf(ss);
    }
  }
}

// H_out = H2 + h[:, :512] ;  V_out = V2 + h[:, 512:] * Vp[..., 512:]
__global__ void __launch_bounds__(256)
final_kernel(const float* __restrict__ H2, const float* __restrict__ hout,
             const float* __restrict__ V2, const float* __restrict__ Vp,
             float* __restrict__ outH, float* __restrict__ outV) {
  long i = (long)blockIdx.x * blockDim.x + threadIdx.x;
  const long total = (long)TOK * 512;
  const long st = (long)gridDim.x * blockDim.x;
  for (; i < total; i += st) {
    const int  j = (int)(i & 511);
    const size_t t = (size_t)(i >> 9);
    outH[i] = H2[i] + hout[t * 1024 + j];
    const float hv = hout[t * 1024 + 512 + j];
#pragma unroll
    for (int c = 0; c < 3; ++c) {
      const size_t vi = (t * 3 + c) * 512 + j;
      outV[vi] = V2[vi] + hv * Vp[(t * 3 + c) * 1024 + 512 + j];
    }
  }
}

// ---------------------------------------------------------------------------
// Host launcher
// ---------------------------------------------------------------------------
extern "C" void kernel_launch(void* const* d_in, const int* in_sizes, int n_in,
                              void* d_out, int out_size, void* d_ws, size_t ws_size,
                              hipStream_t stream) {
  (void)in_sizes; (void)n_in; (void)out_size; (void)ws_size;

  const float* H   = (const float*)d_in[0];
  const float* V   = (const float*)d_in[1];
  const float* Db  = (const float*)d_in[2];
  const float* rbf = (const float*)d_in[3];
  const int*   Hm  = (const int*)  d_in[4];
  const float* g1  = (const float*)d_in[5];
  const float* be1 = (const float*)d_in[6];
  const float* Wq  = (const float*)d_in[7];
  const float* bq  = (const float*)d_in[8];
  const float* Wk  = (const float*)d_in[9];
  const float* bk  = (const float*)d_in[10];
  const float* Wv  = (const float*)d_in[11];
  const float* bv  = (const float*)d_in[12];
  const float* Wvv = (const float*)d_in[13];
  const float* Wo  = (const float*)d_in[14];
  const float* bo  = (const float*)d_in[15];
  const float* Wvo = (const float*)d_in[16];
  const float* g2  = (const float*)d_in[17];
  const float* be2 = (const float*)d_in[18];
  const float* Wlv = (const float*)d_in[19];
  const float* W1  = (const float*)d_in[20];
  const float* b1f = (const float*)d_in[21];
  const float* W2  = (const float*)d_in[22];
  const float* b2f = (const float*)d_in[23];

  char* ws = (char*)d_ws;
  size_t off = 0;
  auto alloc = [&](size_t bytes) -> void* {
    void* p = (void*)(ws + off);
    off = (off + bytes + 255) & ~(size_t)255;
    return p;
  };

  half_t* Hn   = (half_t*)alloc((size_t)TOK  * 512 * 2);
  half_t* Vf16 = (half_t*)alloc((size_t)TOK3 * 512 * 2);
  half_t* WqT  = (half_t*)alloc((size_t)2048 * 512 * 2);
  half_t* WkT  = (half_t*)alloc((size_t)2048 * 512 * 2);
  half_t* WvT  = (half_t*)alloc((size_t)512  * 512 * 2);
  half_t* WvvT = (half_t*)alloc((size_t)512  * 512 * 2);
  half_t* WoT  = (half_t*)alloc((size_t)512  * 512 * 2);
  half_t* WvoT = (half_t*)alloc((size_t)512  * 512 * 2);
  half_t* WlvT = (half_t*)alloc((size_t)1024 * 512 * 2);
  half_t* W1T  = (half_t*)alloc((size_t)2048 * 1024 * 2);
  half_t* W2T  = (half_t*)alloc((size_t)1024 * 2048 * 2);
  half_t* HqA  = (half_t*)alloc((size_t)TOK * 2048 * 2);               // 16 MiB (Vp part 1)
  half_t* HkA  = (half_t*)alloc((size_t)TOK * 2048 * 2);               // 16 MiB (Vp part 2)
  half_t* VtA  = (half_t*)alloc((size_t)4 * NHEAD * 128 * NSEQ * 2);   // 16 MiB (Vp part 3)
  half_t* Hv   = (half_t*)alloc((size_t)TOK  * 512 * 2);               // reused as resH
  half_t* Vv   = (half_t*)alloc((size_t)TOK3 * 512 * 2);               // reused as resV, scaler
  float*  H2   = (float*) alloc((size_t)TOK  * 512 * 4);
  float*  V2f  = (float*) alloc((size_t)TOK3 * 512 * 4);
  half_t* S1   = (half_t*)alloc((size_t)TOK * 2048 * 2);
  float*  hout = (float*) alloc((size_t)TOK * 1024 * 4);

  // lifetime-based aliases (all sizes are 256B multiples -> regions contiguous)
  float*  Vp     = (float*)HqA;   // TOK3 x 1024 f32 = 48 MiB over HqA+HkA+VtA
  half_t* resH   = Hv;
  half_t* resV   = Vv;
  half_t* V2h    = Vf16;
  half_t* Hn2    = Hn;
  half_t* scaler = Vv;

  auto gemm = [&](const half_t* A, int lda, const half_t* Bt, int ldb,
                  const float* bias, const float* resid, int ldr,
                  float* Cf, int ldcf, half_t* Ch, int ldch,
                  int M, int N, int K, int act) {
    dim3 grid(N / 128, M / 128);
    gemm_wmma_kernel<<<grid, 256, 0, stream>>>(A, lda, Bt, ldb, bias, resid, ldr,
                                               Cf, ldcf, Ch, ldch, M, N, K, act);
  };
  auto tcast = [&](const float* W, half_t* Wt, int K, int N) {
    long n = (long)K * N;
    transpose_cast_kernel<<<(int)((n + 255) / 256), 256, 0, stream>>>(W, Wt, K, N);
  };

  // Phase A: normalize / cast / transpose weights
  layernorm_kernel<<<TOK / 8, 256, 0, stream>>>(H, g1, be1, Hn, 512);
  cast_kernel<<<(TOK3 * 512 + 255) / 256, 256, 0, stream>>>(V, Vf16, (long)TOK3 * 512);
  tcast(Wq,  WqT,  512, 2048);
  tcast(Wk,  WkT,  512, 2048);
  tcast(Wv,  WvT,  512, 512);
  tcast(Wvv, WvvT, 512, 512);
  tcast(Wo,  WoT,  512, 512);
  tcast(Wvo, WvoT, 512, 512);
  tcast(Wlv, WlvT, 512, 1024);
  tcast(W1,  W1T,  1024, 2048);
  tcast(W2,  W2T,  2048, 1024);

  // Phase B: projections
  gemm(Hn, 512, WqT, 512, bq, nullptr, 0, nullptr, 0, HqA, 2048, TOK, 2048, 512, 0);
  gemm(Hn, 512, WkT, 512, bk, nullptr, 0, nullptr, 0, HkA, 2048, TOK, 2048, 512, 0);
  gemm(Hn, 512, WvT, 512, bv, nullptr, 0, nullptr, 0, Hv,  512,  TOK, 512,  512, 0);
  gemm(Vf16, 512, WvvT, 512, nullptr, nullptr, 0, nullptr, 0, Vv, 512, TOK3, 512, 512, 0);

  // Phase C: pack V_attn^T, then flash attention
  pack_vt_kernel<<<32768, 256, 0, stream>>>(Hv, Vv, VtA);
  attn_kernel<<<4 * NHEAD * (NSEQ / 128), 256, 0, stream>>>(HqA, HkA, VtA, rbf, Db,
                                                            Hm, resH, resV);

  // Phase D: output projections with residuals
  gemm(resH, 512, WoT,  512, bo, H, 512, H2,  512, nullptr, 0, TOK,  512, 512, 0);
  gemm(resV, 512, WvoT, 512, nullptr, V, 512, V2f, 512, V2h, 512, TOK3, 512, 512, 0);

  // Phase E: gated FFN
  layernorm_kernel<<<TOK / 8, 256, 0, stream>>>(H2, g2, be2, Hn2, 512);
  gemm(V2h, 512, WlvT, 512, nullptr, nullptr, 0, Vp, 1024, nullptr, 0, TOK3, 1024, 512, 0);
  pack_scaler_kernel<<<16384, 256, 0, stream>>>(Hn2, Vp, scaler);
  gemm(scaler, 1024, W1T, 1024, b1f, nullptr, 0, nullptr, 0, S1, 2048, TOK, 2048, 1024, 1);
  gemm(S1, 2048, W2T, 2048, b2f, nullptr, 0, hout, 1024, nullptr, 0, TOK, 1024, 2048, 0);

  // Phase F: final residual combine into d_out (H then V)
  float* outH = (float*)d_out;
  float* outV = outH + (size_t)TOK * 512;
  final_kernel<<<8192, 256, 0, stream>>>(H2, hout, V2f, Vp, outH, outV);
}
